// GraphVampNet_13907104104634
// MI455X (gfx1250) — compile-verified, matrix operand order checked by reading, not run on hardware
//
#include <hip/hip_runtime.h>
#include <hip/hip_bf16.h>

// ---------------------------------------------------------------------------
// GraphVampNet on MI455X (gfx1250, wave32, WMMA)
//
// Restructured math (per layer l):
//   ST = atom_emb @ [W_self | W_nbr]          (bf16x3 WMMA GEMM, [16384,256]x[256,1024])
//   z[b,n,m,o] = ST[bn][o] + ST[b*256+idx][512+o] + gauss(d) . W_bond[:,o] + bias[o]
//   BN1 via deterministic 2-level reduction, folded to scale/shift
//   nbr_sumed = sum_m sigmoid(zf)*relu(zc);  BN2;  atom_emb = relu(atom_emb + .)
// Final: pool over N, 256->6 linear, softmax.
//
// GEMM data movement: the 64-col weight strip (hi+lo bf16, 64 KB) is staged
// into LDS once per workgroup via CDNA5 async global->LDS copies
// (GLOBAL_LOAD_ASYNC_TO_LDS_B128, ASYNCcnt) and read back with ds_load_b128,
// removing the 8x per-wave duplication of B-operand global loads.
// ---------------------------------------------------------------------------

typedef __bf16 v16bf __attribute__((ext_vector_type(16)));
typedef float  v8f   __attribute__((ext_vector_type(8)));
typedef int    v4i_g __attribute__((vector_size(16)));   // matches builtin param

#if defined(__has_builtin)
#  if __has_builtin(__builtin_amdgcn_global_load_async_to_lds_b128) && \
      __has_builtin(__builtin_amdgcn_s_wait_asynccnt)
#    define USE_ASYNC_LDS 1
#  endif
#endif
#ifndef USE_ASYNC_LDS
#  define USE_ASYNC_LDS 0
#endif

#define KB 1024ull
#define MB (1024ull * KB)

// workspace layout (bytes)
static const size_t OFF_ATOM = 0;              // [64*256*256] f32  = 16 MB
static const size_t OFF_ST   = 16 * MB;        // [16384*1024] f32  = 64 MB
static const size_t OFF_NS   = 80 * MB;        // [16384*256]  f32  = 16 MB
static const size_t OFF_AH   = 96 * MB;        // [16384*256]  bf16 =  8 MB
static const size_t OFF_AL   = 104 * MB;       // [16384*256]  bf16 =  8 MB
static const size_t OFF_WH   = 112 * MB;       // [1024*256]   bf16 = 512 KB
static const size_t OFF_WL   = 112 * MB + 512 * KB;
static const size_t OFF_PS   = 113 * MB;       // [1024*512]   f32  =  2 MB
static const size_t OFF_PQ   = 115 * MB;       // [1024*512]   f32  =  2 MB
static const size_t OFF_SC1  = 117 * MB;       // [512] f32
static const size_t OFF_SH1  = OFF_SC1 + 4 * KB;
static const size_t OFF_SC2  = OFF_SC1 + 8 * KB;
static const size_t OFF_SH2  = OFF_SC1 + 12 * KB;

// ---------------------------------------------------------------------------
// helpers
// ---------------------------------------------------------------------------
__device__ __forceinline__ v16bf load_k32(const __bf16* base) {
  // ISA 16-bit operand layout: per lane, elements 0..7 = K+0..7,
  // elements 8..15 = K+16..23 (lane-half adds +8 to K, handled by caller).
  union { v16bf v; uint4 q[2]; } u;
  u.q[0] = *(const uint4*)(base);
  u.q[1] = *(const uint4*)(base + 16);
  return u.v;
}

// ---------------------------------------------------------------------------
// atom_emb init: broadcast emb_table over batch
// ---------------------------------------------------------------------------
__global__ void k_init_emb(const float* __restrict__ emb, float* __restrict__ atom) {
  int i = blockIdx.x * 256 + threadIdx.x;      // 4194304
  atom[i] = emb[i & 65535];                    // (n*256 + h)
}

// ---------------------------------------------------------------------------
// split f32 -> bf16 hi/lo pair (bf16x3 trick)
// ---------------------------------------------------------------------------
__global__ void k_split(const float* __restrict__ x, __bf16* __restrict__ hi,
                        __bf16* __restrict__ lo) {
  int i = blockIdx.x * 256 + threadIdx.x;
  float v = x[i];
  __bf16 h = (__bf16)v;
  hi[i] = h;
  lo[i] = (__bf16)(v - (float)h);
}

// build combined weight, transposed [n=1024][k=256]: n<512 -> W_self, else W_nbr
__global__ void k_buildW(const float* __restrict__ W, __bf16* __restrict__ hi,
                         __bf16* __restrict__ lo) {
  int i = blockIdx.x * 256 + threadIdx.x;      // 262144
  int n = i >> 8, k = i & 255;
  float v = (n < 512) ? W[k * 512 + n] : W[(256 + k) * 512 + (n - 512)];
  __bf16 h = (__bf16)v;
  hi[i] = h;
  lo[i] = (__bf16)(v - (float)h);
}

// ---------------------------------------------------------------------------
// bf16x3 WMMA GEMM:  ST[16384,1024] = A[16384,256] @ Wt^T   (Wt is [1024,256])
// grid = 2048 (128 m-blocks x 16 n-blocks), block = 256 (8 waves)
// B strip staged in LDS per WG (async copy); each wave: one 16x64 strip.
// ---------------------------------------------------------------------------
__global__ __launch_bounds__(256) void k_gemm(
    const __bf16* __restrict__ Ah, const __bf16* __restrict__ Al,
    const __bf16* __restrict__ Bh, const __bf16* __restrict__ Bl,
    float* __restrict__ ST) {
  __shared__ __bf16 sB[2 * 64 * 256];          // [hi|lo][col][k] = 64 KB
  const int t    = threadIdx.x;
  const int mblk = blockIdx.x >> 4;
  const int nblk = blockIdx.x & 15;

  // ---- stage the 64-column weight strip (hi then lo) into LDS -------------
  const __bf16* gBh = Bh + nblk * 64 * 256;    // 32 KB contiguous
  const __bf16* gBl = Bl + nblk * 64 * 256;
#if USE_ASYNC_LDS
#pragma unroll
  for (int i = 0; i < 8; ++i) {
    const int c = i * 256 + t;                 // 16-byte chunk id (2048/half)
    __builtin_amdgcn_global_load_async_to_lds_b128(
        (__attribute__((address_space(1))) v4i_g*)(gBh + c * 8),
        (__attribute__((address_space(3))) v4i_g*)(sB + c * 8), 0, 0);
    __builtin_amdgcn_global_load_async_to_lds_b128(
        (__attribute__((address_space(1))) v4i_g*)(gBl + c * 8),
        (__attribute__((address_space(3))) v4i_g*)(sB + 16384 + c * 8), 0, 0);
  }
  __builtin_amdgcn_s_wait_asynccnt(0);
  __syncthreads();
#else
  {
    const uint4* srcH = (const uint4*)gBh;
    const uint4* srcL = (const uint4*)gBl;
    uint4* dst = (uint4*)sB;
#pragma unroll
    for (int i = 0; i < 8; ++i) {
      const int c = i * 256 + t;
      dst[c]        = srcH[c];
      dst[2048 + c] = srcL[c];
    }
    __syncthreads();
  }
#endif

  // ---- compute ------------------------------------------------------------
  const int lane  = t & 31;
  const int wv    = t >> 5;
  const int half  = lane >> 4;
  const int r     = lane & 15;
  const int mtile = mblk * 8 + wv;

  const __bf16* pah = Ah + (mtile * 16 + r) * 256;
  const __bf16* pal = Al + (mtile * 16 + r) * 256;

  v8f acc[4] = {v8f{}, v8f{}, v8f{}, v8f{}};

  for (int kk = 0; kk < 256; kk += 32) {
    const int koff = kk + 8 * half;
    v16bf ah = load_k32(pah + koff);
    v16bf al = load_k32(pal + koff);
#pragma unroll
    for (int nt = 0; nt < 4; ++nt) {
      const __bf16* pb = sB + (nt * 16 + r) * 256 + koff;   // LDS (ds_load)
      v16bf bh = load_k32(pb);
      v16bf bl = load_k32(pb + 16384);
      acc[nt] = __builtin_amdgcn_wmma_f32_16x16x32_bf16(
          false, ah, false, bh, (short)0, acc[nt], false, false);
      acc[nt] = __builtin_amdgcn_wmma_f32_16x16x32_bf16(
          false, ah, false, bl, (short)0, acc[nt], false, false);
      acc[nt] = __builtin_amdgcn_wmma_f32_16x16x32_bf16(
          false, al, false, bh, (short)0, acc[nt], false, false);
    }
  }

#pragma unroll
  for (int nt = 0; nt < 4; ++nt) {
    const int ncol = nblk * 64 + nt * 16 + r;
#pragma unroll
    for (int j = 0; j < 8; ++j) {
      int mrow = mtile * 16 + j + 8 * half;   // C/D layout: VGPR j, lane-half
      ST[mrow * 1024 + ncol] = acc[nt][j];
    }
  }
}

// ---------------------------------------------------------------------------
// BN1 stats pass: per-WG partial sum/sumsq over 16 nodes x 16 m per channel
// grid = 1024, block = 256 (thread t owns channels t and t+256)
// ---------------------------------------------------------------------------
__global__ __launch_bounds__(256) void k_stats1(
    const float* __restrict__ ST, const int* __restrict__ idx,
    const float* __restrict__ dist, const float* __restrict__ Wl,
    const float* __restrict__ bias, float* __restrict__ ps, float* __restrict__ pq) {
  __shared__ float sWb[16 * 512];  // bond weights, 32 KB
  __shared__ float sBias[512];
  __shared__ float sG[256];        // 16 m x 16 filters for current node
  const int t = threadIdx.x;
  for (int j = t; j < 8192; j += 256) {
    int f = j >> 9, o = j & 511;
    sWb[j] = Wl[(512 + f) * 512 + o];
  }
  for (int j = t; j < 512; j += 256) sBias[j] = bias[j];
  __syncthreads();

  const int o1 = t, o2 = t + 256;
  float s1 = 0.f, q1 = 0.f, s2 = 0.f, q2 = 0.f;
  const int node0 = blockIdx.x * 16;
  for (int nd = 0; nd < 16; ++nd) {
    const int node = node0 + nd;
    const int b = node >> 8;
    {
      int m = t >> 4, f = t & 15;
      float d = dist[node * 16 + m] - 0.2f * (float)f;
      sG[t] = __expf(-d * d * 25.0f);            // / STEP^2 = *25
    }
    __syncthreads();
    const float* selfrow = ST + node * 1024;
    const float sf1 = selfrow[o1] + sBias[o1];
    const float sf2 = selfrow[o2] + sBias[o2];
    for (int m = 0; m < 16; ++m) {
      const int nb = idx[node * 16 + m];
      const float* nrow = ST + (b * 256 + nb) * 1024 + 512;
      float z1 = sf1 + nrow[o1];
      float z2 = sf2 + nrow[o2];
#pragma unroll
      for (int f = 0; f < 16; ++f) {
        float g = sG[m * 16 + f];
        z1 = fmaf(g, sWb[f * 512 + o1], z1);
        z2 = fmaf(g, sWb[f * 512 + o2], z2);
      }
      s1 += z1; q1 = fmaf(z1, z1, q1);
      s2 += z2; q2 = fmaf(z2, z2, q2);
    }
    __syncthreads();
  }
  ps[blockIdx.x * 512 + o1] = s1; ps[blockIdx.x * 512 + o2] = s2;
  pq[blockIdx.x * 512 + o1] = q1; pq[blockIdx.x * 512 + o2] = q2;
}

// finalize BN1: fixed-order sum of 1024 partials per channel -> scale/shift
__global__ void k_finstats1(const float* __restrict__ ps, const float* __restrict__ pq,
                            const float* __restrict__ gh, const float* __restrict__ bh,
                            float* __restrict__ scale1, float* __restrict__ shift1) {
  int o = blockIdx.x * 256 + threadIdx.x;      // 0..511
  float s = 0.f, q = 0.f;
  for (int g = 0; g < 1024; ++g) { s += ps[g * 512 + o]; q += pq[g * 512 + o]; }
  const float inv = 1.0f / 262144.0f;
  float mu  = s * inv;
  float var = q * inv - mu * mu;
  float rs  = rsqrtf(var + 1e-5f);
  float sc  = rs * gh[o];
  scale1[o] = sc;
  shift1[o] = bh[o] - mu * sc;
}

// ---------------------------------------------------------------------------
// pass 2: recompute z, apply BN1, gated aggregation over m -> nbr_sumed
// grid = 16384 (one node), block = 256 (thread t: filter ch t, core ch t+256)
// ---------------------------------------------------------------------------
__global__ __launch_bounds__(256) void k_pass2(
    const float* __restrict__ ST, const int* __restrict__ idx,
    const float* __restrict__ dist, const float* __restrict__ Wl,
    const float* __restrict__ bias, const float* __restrict__ scale1,
    const float* __restrict__ shift1, float* __restrict__ ns) {
  __shared__ float sWb[16 * 512];
  __shared__ float sG[256];
  __shared__ float sSc[512], sSh[512];
  const int t = threadIdx.x;
  for (int j = t; j < 8192; j += 256) {
    int f = j >> 9, o = j & 511;
    sWb[j] = Wl[(512 + f) * 512 + o];
  }
  for (int j = t; j < 512; j += 256) { sSc[j] = scale1[j]; sSh[j] = shift1[j]; }
  const int node = blockIdx.x;
  const int b = node >> 8;
  {
    int m = t >> 4, f = t & 15;
    float d = dist[node * 16 + m] - 0.2f * (float)f;
    sG[t] = __expf(-d * d * 25.0f);
  }
  __syncthreads();

  const int of = t, oc = t + 256;
  const float* selfrow = ST + node * 1024;
  const float sf = selfrow[of] + bias[of];
  const float sc = selfrow[oc] + bias[oc];
  float acc = 0.0f;
  for (int m = 0; m < 16; ++m) {
    const int nb = idx[node * 16 + m];
    const float* nrow = ST + (b * 256 + nb) * 1024 + 512;
    float zf = sf + nrow[of];
    float zc = sc + nrow[oc];
#pragma unroll
    for (int f = 0; f < 16; ++f) {
      float g = sG[m * 16 + f];
      zf = fmaf(g, sWb[f * 512 + of], zf);
      zc = fmaf(g, sWb[f * 512 + oc], zc);
    }
    zf = fmaf(zf, sSc[of], sSh[of]);
    zc = fmaf(zc, sSc[oc], sSh[oc]);
    float sig = 1.0f / (1.0f + __expf(-zf));
    acc = fmaf(sig, fmaxf(zc, 0.0f), acc);
  }
  ns[node * 256 + t] = acc;
}

// BN2 stats: one WG per channel, deterministic tree reduce over 16384 rows
__global__ __launch_bounds__(256) void k_stats2(
    const float* __restrict__ ns, const float* __restrict__ go,
    const float* __restrict__ bo, float* __restrict__ scale2,
    float* __restrict__ shift2) {
  __shared__ float rs_[256], rq_[256];
  const int o = blockIdx.x, t = threadIdx.x;
  float s = 0.f, q = 0.f;
  for (int row = t; row < 16384; row += 256) {
    float v = ns[row * 256 + o];
    s += v; q = fmaf(v, v, q);
  }
  rs_[t] = s; rq_[t] = q;
  __syncthreads();
  for (int st = 128; st > 0; st >>= 1) {
    if (t < st) { rs_[t] += rs_[t + st]; rq_[t] += rq_[t + st]; }
    __syncthreads();
  }
  if (t == 0) {
    const float inv = 1.0f / 16384.0f;
    float mu  = rs_[0] * inv;
    float var = rq_[0] * inv - mu * mu;
    float r   = rsqrtf(var + 1e-5f);
    float sc  = r * go[o];
    scale2[o] = sc;
    shift2[o] = bo[o] - mu * sc;
  }
}

// residual + relu update:  atom = relu(atom + bn2(ns))
__global__ void k_update(float* __restrict__ atom, const float* __restrict__ ns,
                         const float* __restrict__ scale2,
                         const float* __restrict__ shift2) {
  int i = blockIdx.x * 256 + threadIdx.x;      // block == row
  int o = threadIdx.x;
  float v = atom[i] + fmaf(ns[i], scale2[o], shift2[o]);
  atom[i] = fmaxf(v, 0.0f);
}

// pooling + classifier + softmax, one WG per batch element
__global__ __launch_bounds__(256) void k_final(
    const float* __restrict__ atom, const float* __restrict__ Wc,
    const float* __restrict__ bc, float* __restrict__ out) {
  __shared__ float pooled[256];
  __shared__ float logits[6];
  const int b = blockIdx.x, h = threadIdx.x;
  const float* base = atom + b * 65536;
  float s = 0.f;
  for (int n = 0; n < 256; ++n) s += fmaxf(base[n * 256 + h], 0.0f);
  pooled[h] = s * (1.0f / 256.0f);
  __syncthreads();
  if (h < 6) {
    float l = bc[h];
    for (int k = 0; k < 256; ++k) l = fmaf(pooled[k], Wc[k * 6 + h], l);
    logits[h] = l;
  }
  __syncthreads();
  if (h == 0) {
    float mx = logits[0];
    for (int c = 1; c < 6; ++c) mx = fmaxf(mx, logits[c]);
    float e[6], sum = 0.f;
    for (int c = 0; c < 6; ++c) { e[c] = __expf(logits[c] - mx); sum += e[c]; }
    float inv = 1.0f / sum;
    for (int c = 0; c < 6; ++c) out[b * 6 + c] = e[c] * inv;
  }
}

// ---------------------------------------------------------------------------
extern "C" void kernel_launch(void* const* d_in, const int* in_sizes, int n_in,
                              void* d_out, int out_size, void* d_ws, size_t ws_size,
                              hipStream_t stream) {
  const float* d_dist = (const float*)d_in[0];
  const int*   d_idx  = (const int*)d_in[1];
  const float* d_emb  = (const float*)d_in[2];
  const float* d_W    = (const float*)d_in[3];
  const float* d_b    = (const float*)d_in[4];
  const float* d_gh   = (const float*)d_in[5];
  const float* d_bh   = (const float*)d_in[6];
  const float* d_go   = (const float*)d_in[7];
  const float* d_bo   = (const float*)d_in[8];
  const float* d_Wc   = (const float*)d_in[9];
  const float* d_bc   = (const float*)d_in[10];

  char* ws = (char*)d_ws;
  float*  atom = (float*)(ws + OFF_ATOM);
  float*  ST   = (float*)(ws + OFF_ST);
  float*  ns   = (float*)(ws + OFF_NS);
  __bf16* Ah   = (__bf16*)(ws + OFF_AH);
  __bf16* Al   = (__bf16*)(ws + OFF_AL);
  __bf16* Wh   = (__bf16*)(ws + OFF_WH);
  __bf16* Wl   = (__bf16*)(ws + OFF_WL);
  float*  ps   = (float*)(ws + OFF_PS);
  float*  pq   = (float*)(ws + OFF_PQ);
  float*  sc1  = (float*)(ws + OFF_SC1);
  float*  sh1  = (float*)(ws + OFF_SH1);
  float*  sc2  = (float*)(ws + OFF_SC2);
  float*  sh2  = (float*)(ws + OFF_SH2);

  k_init_emb<<<16384, 256, 0, stream>>>(d_emb, atom);

  for (int l = 0; l < 3; ++l) {
    const float* Wlay = d_W + (size_t)l * 528 * 512;
    const float* blay = d_b + l * 512;
    k_split<<<16384, 256, 0, stream>>>(atom, Ah, Al);
    k_buildW<<<1024, 256, 0, stream>>>(Wlay, Wh, Wl);
    k_gemm<<<2048, 256, 0, stream>>>(Ah, Al, Wh, Wl, ST);
    k_stats1<<<1024, 256, 0, stream>>>(ST, d_idx, d_dist, Wlay, blay, ps, pq);
    k_finstats1<<<2, 256, 0, stream>>>(ps, pq, d_gh + l * 512, d_bh + l * 512, sc1, sh1);
    k_pass2<<<16384, 256, 0, stream>>>(ST, d_idx, d_dist, Wlay, blay, sc1, sh1, ns);
    k_stats2<<<256, 256, 0, stream>>>(ns, d_go + l * 256, d_bo + l * 256, sc2, sh2);
    k_update<<<16384, 256, 0, stream>>>(atom, ns, sc2, sh2);
  }

  k_final<<<64, 256, 0, stream>>>(atom, d_Wc, d_bc, (float*)d_out);
}
